// ContrastiveLoss_51823075393886
// MI455X (gfx1250) — matrix operand order repaired and checked by reading, not compiled, not run on hardware
//
#include <hip/hip_runtime.h>
#include <hip/hip_bf16.h>
#include <math.h>

// ---------------------------------------------------------------------------
// SCAN t2i contrastive loss on gfx1250 (MI455X), wave32 + WMMA bf16.
// One workgroup (128 threads = 4 waves) per (caption c, image i) pair:
//   GEMM1: attn[36x24] = images[i](36x512) x captions[c]^T(512x24)  via
//          v_wmma_f32_16x16x32_bf16 (K unrolled x4), tiles kept in LDS.
//   Fused: leakyrelu+mask -> L2 norm over w -> softmax over r -> s^T (bf16).
//   GEMM2: wei[24x512] = s^T(24x36) x images[i](36x512), accumulators folded
//          directly into per-word cap.wei and ||wei||^2 (wei never stored).
//   logsumexp over valid words -> scores[i][c].
// Final kernel: margin hinge max over rows/cols -> scalar loss.
// ---------------------------------------------------------------------------

typedef __attribute__((ext_vector_type(16))) __bf16 v16bf;
typedef __attribute__((ext_vector_type(8)))  float  v8f;

#define I_N   128
#define R_N   36
#define R_PAD 48
#define W_N   24
#define W_PAD 32
#define D_N   512
#define K_R   64          // padded r-extent used as K of GEMM2
#define MARGIN 0.2f
#define LAM_LSE 6.0f
#define LAM_SM  9.0f
#define EPSV 1e-8f

__device__ __forceinline__ unsigned short f2bf(float f) {
  unsigned int x = __float_as_uint(f);
  unsigned int r = x + 0x7FFFu + ((x >> 16) & 1u);   // round-to-nearest-even
  return (unsigned short)(r >> 16);
}

// Build one 16x32 bf16 A fragment + one 32x16 bf16 B fragment for the
// wave32 V_WMMA_F32_16X16X32_BF16 VGPR layouts (ISA 7.12.2) and accumulate.
__device__ __forceinline__ v8f wmma_step(const __bf16* __restrict__ arow,
                                         const __bf16* __restrict__ brow,
                                         int kb, int koffA, int hi, v8f acc) {
  v16bf a, b;
#pragma unroll
  for (int j = 0; j < 8; ++j) {
    const int ka = (j < 4) ? (kb + koffA + 2 * j)
                           : (kb + 16 + koffA + 2 * (j - 4));
    a[2 * j]     = arow[ka];
    a[2 * j + 1] = arow[ka + 1];
    const int kbb = (j < 4) ? (kb + hi * 8 + 2 * j)
                            : (kb + 16 + hi * 8 + 2 * (j - 4));
    b[2 * j]     = brow[kbb];
    b[2 * j + 1] = brow[kbb + 1];
  }
  return __builtin_amdgcn_wmma_f32_16x16x32_bf16(
      false, a, false, b, (short)0, acc, false, false);
}

// ----------------------------- prep kernels --------------------------------

// images (I,R,D) f32 -> bf16 padded to (I, R_PAD, D)
__global__ void prep_img_bf16(const float* __restrict__ img,
                              unsigned short* __restrict__ out) {
  long long idx = (long long)blockIdx.x * blockDim.x + threadIdx.x;
  if (idx >= (long long)I_N * R_PAD * D_N) return;
  int d = (int)(idx % D_N);
  long long t = idx / D_N;
  int r = (int)(t % R_PAD);
  int i = (int)(t / R_PAD);
  float v = (r < R_N) ? img[((long long)i * R_N + r) * D_N + d] : 0.0f;
  out[idx] = f2bf(v);
}

// images -> bf16 transposed (I, D, K_R) with r zero-padded to K_R
__global__ void prep_imgT_bf16(const float* __restrict__ img,
                               unsigned short* __restrict__ out) {
  long long idx = (long long)blockIdx.x * blockDim.x + threadIdx.x;
  if (idx >= (long long)I_N * D_N * K_R) return;
  int r = (int)(idx % K_R);
  long long t = idx / K_R;
  int d = (int)(t % D_N);
  int i = (int)(t / D_N);
  float v = (r < R_N) ? img[((long long)i * R_N + r) * D_N + d] : 0.0f;
  out[idx] = f2bf(v);
}

// captions (C,W,D) f32 -> bf16 padded to (C, W_PAD, D)
__global__ void prep_cap_bf16(const float* __restrict__ cap,
                              unsigned short* __restrict__ out) {
  long long idx = (long long)blockIdx.x * blockDim.x + threadIdx.x;
  if (idx >= (long long)I_N * W_PAD * D_N) return;
  int d = (int)(idx % D_N);
  long long t = idx / D_N;
  int w = (int)(t % W_PAD);
  int c = (int)(t / W_PAD);
  float v = (w < W_N) ? cap[((long long)c * W_N + w) * D_N + d] : 0.0f;
  out[idx] = f2bf(v);
}

// per-(c,w) caption L2 norm
__global__ void prep_capnorm(const float* __restrict__ cap,
                             float* __restrict__ capnorm) {
  int c = blockIdx.x;
  int w = threadIdx.x;
  if (w >= W_N) return;
  const float* row = cap + ((long long)c * W_N + w) * D_N;
  float s = 0.0f;
  for (int d = 0; d < D_N; ++d) { float v = row[d]; s += v * v; }
  capnorm[c * W_N + w] = sqrtf(s);
}

// ------------------------------ main kernel --------------------------------

__global__ void __launch_bounds__(128)
scan_t2i_pair_kernel(const unsigned short* __restrict__ ws_img,   // (I,R_PAD,D) bf16
                     const unsigned short* __restrict__ ws_imgT,  // (I,D,K_R)   bf16
                     const unsigned short* __restrict__ ws_cap,   // (C,W_PAD,D) bf16
                     const float* __restrict__ captions,          // (C,W,D)     f32
                     const float* __restrict__ capnorm,           // (C,W)
                     const int*   __restrict__ cap_lens,          // (C,)
                     float* __restrict__ scores) {                // (I,C)
  const int c = blockIdx.x;
  const int i = blockIdx.y;

  __shared__ float          s_a[R_PAD][W_PAD + 1];   // attn -> s (padded stride)
  __shared__ unsigned short s_sT[W_PAD][K_R];        // s^T in bf16 (A of GEMM2)
  __shared__ float          s_num[W_PAD];
  __shared__ float          s_nrm[W_PAD];
  __shared__ int            s_len;

  const int tid   = threadIdx.x;
  const int lane  = tid & 31;
  const int wv    = tid >> 5;          // wave id 0..3
  const int hi    = lane >> 4;         // lane half
  const int ln    = lane & 15;
  const int koffA = hi * 8;            // A-fragment K sub-offset (16-bit layout)

  if (tid == 0) s_len = cap_lens[c];
  if (tid < W_PAD) { s_num[tid] = 0.0f; s_nrm[tid] = 0.0f; }

  // ---- Phase 1: attn = images[i] x captions[c]^T  (bf16 WMMA, f32 acc) ----
  {
    const unsigned short* imgb = ws_img + (long long)i * R_PAD * D_N;
    const unsigned short* capb = ws_cap + (long long)c * W_PAD * D_N;
    for (int t = wv; t < 6; t += 4) {
      const int mt = t >> 1, nt = t & 1;
      const __bf16* arow = (const __bf16*)(imgb + (long long)(mt * 16 + ln) * D_N);
      const __bf16* brow = (const __bf16*)(capb + (long long)(nt * 16 + ln) * D_N);
      v8f acc = {0.f, 0.f, 0.f, 0.f, 0.f, 0.f, 0.f, 0.f};
#pragma unroll 4
      for (int kb = 0; kb < D_N; kb += 32) {
        if ((kb & 127) == 0) {                         // one prefetch / 128 K
          __builtin_prefetch(arow + kb + 128, 0, 1);   // global_prefetch hint
          __builtin_prefetch(brow + kb + 128, 0, 1);
        }
        acc = wmma_step(arow, brow, kb, koffA, hi, acc);
      }
#pragma unroll
      for (int p = 0; p < 8; ++p)
        s_a[mt * 16 + p + hi * 8][nt * 16 + ln] = acc[p];
    }
  }
  __syncthreads();

  // ---- Phase 2a: leaky-relu * word mask -----------------------------------
  const int len = s_len;
  for (int idx = tid; idx < R_PAD * W_PAD; idx += 128) {
    const int r = idx / W_PAD, w = idx % W_PAD;
    float v = s_a[r][w];
    v = (v > 0.0f) ? v : 0.1f * v;
    if (r >= R_N || w >= W_N || w >= len) v = 0.0f;
    s_a[r][w] = v;
  }
  __syncthreads();

  // ---- Phase 2b: L2 normalize over w (per r) ------------------------------
  if (tid < R_PAD) {
    float ss = 0.0f;
    for (int w = 0; w < W_PAD; ++w) { float v = s_a[tid][w]; ss += v * v; }
    const float inv = 1.0f / (sqrtf(ss) + EPSV);
    for (int w = 0; w < W_PAD; ++w) s_a[tid][w] *= inv;
  }
  __syncthreads();

  // ---- Phase 2c: softmax over r (per w), lambda=9 -------------------------
  if (tid < W_PAD) {
    const int w = tid;
    float m = -3.0e38f;
    for (int r = 0; r < R_N; ++r) m = fmaxf(m, LAM_SM * s_a[r][w]);
    float sum = 0.0f;
    for (int r = 0; r < R_N; ++r) sum += __expf(LAM_SM * s_a[r][w] - m);
    const float inv = 1.0f / sum;
    for (int r = 0; r < R_N; ++r)
      s_a[r][w] = __expf(LAM_SM * s_a[r][w] - m) * inv;
  }
  __syncthreads();

  // ---- Phase 2d: s^T in bf16 (A operand of GEMM2), K=r padded to 64 -------
  for (int idx = tid; idx < W_PAD * K_R; idx += 128) {
    const int w = idx >> 6, r = idx & 63;
    const float v = (w < W_N && r < R_N) ? s_a[r][w] : 0.0f;
    s_sT[w][r] = f2bf(v);
  }
  __syncthreads();

  // ---- Phase 3: wei = s^T x images[i], fused with cos accumulation --------
  {
    const unsigned short* imgT = ws_imgT + (long long)i * D_N * K_R;
    float numAcc[16], nrmAcc[16];
#pragma unroll
    for (int q = 0; q < 16; ++q) { numAcc[q] = 0.0f; nrmAcc[q] = 0.0f; }

    for (int idx = wv; idx < 64; idx += 4) {
      const int mt = idx & 1, nt = idx >> 1;
      const __bf16* arow = (const __bf16*)&s_sT[mt * 16 + ln][0];
      const __bf16* brow = (const __bf16*)(imgT + (long long)(nt * 16 + ln) * K_R);
      v8f acc = {0.f, 0.f, 0.f, 0.f, 0.f, 0.f, 0.f, 0.f};
      acc = wmma_step(arow, brow, 0,  koffA, hi, acc);
      acc = wmma_step(arow, brow, 32, koffA, hi, acc);

      const int d0 = nt * 16 + ln;
#pragma unroll
      for (int p = 0; p < 8; ++p) {
        const int w = mt * 16 + p + hi * 8;
        if (w < W_N) {
          const float wei = acc[p];
          const float cv  = captions[((long long)c * W_N + w) * D_N + d0];
          numAcc[mt * 8 + p] += cv * wei;
          nrmAcc[mt * 8 + p] += wei * wei;
        }
      }
    }
#pragma unroll
    for (int mt = 0; mt < 2; ++mt)
#pragma unroll
      for (int p = 0; p < 8; ++p) {
        const int w = mt * 16 + p + hi * 8;
        if (w < W_N) {
          atomicAdd(&s_num[w], numAcc[mt * 8 + p]);   // ds_add_f32
          atomicAdd(&s_nrm[w], nrmAcc[mt * 8 + p]);
        }
      }
  }
  __syncthreads();

  // ---- Phase 4: per-word cosine + logsumexp over valid words --------------
  if (tid == 0) {
    float zmax = -3.0e38f;
    float zv[W_N];
    for (int w = 0; w < W_N; ++w) {
      float z = -3.0e38f;
      if (w < len) {
        const float den = fmaxf(capnorm[c * W_N + w] * sqrtf(s_nrm[w]), EPSV);
        z = LAM_LSE * (s_num[w] / den);
      }
      zv[w] = z;
      zmax = fmaxf(zmax, z);
    }
    float sum = 0.0f;
    for (int w = 0; w < W_N; ++w)
      if (w < len) sum += __expf(zv[w] - zmax);
    scores[(long long)i * I_N + c] = (zmax + __logf(sum)) / LAM_LSE;
  }
}

// ------------------------- final hinge-loss kernel -------------------------

__global__ void __launch_bounds__(128)
hinge_loss_kernel(const float* __restrict__ sc, float* __restrict__ out) {
  __shared__ float red[128];
  const int t = threadIdx.x;
  const float diag_t = sc[t * I_N + t];
  float rm = 0.0f, cm = 0.0f;
  for (int k = 0; k < I_N; ++k) {
    if (k != t) {
      rm = fmaxf(rm, fmaxf(0.0f, MARGIN + sc[t * I_N + k] - diag_t)); // row max (cost_s)
      cm = fmaxf(cm, fmaxf(0.0f, MARGIN + sc[k * I_N + t] - diag_t)); // col max (cost_im)
    }
  }
  red[t] = rm + cm;
  __syncthreads();
  for (int s = 64; s > 0; s >>= 1) {
    if (t < s) red[t] += red[t + s];
    __syncthreads();
  }
  if (t == 0) out[0] = red[0];
}

// ------------------------------- launcher ----------------------------------

extern "C" void kernel_launch(void* const* d_in, const int* in_sizes, int n_in,
                              void* d_out, int out_size, void* d_ws, size_t ws_size,
                              hipStream_t stream) {
  const float* images   = (const float*)d_in[0];   // (128,36,512)
  const float* captions = (const float*)d_in[1];   // (128,24,512)
  const int*   cap_lens = (const int*)d_in[2];     // (128,)

  // workspace layout
  char* base = (char*)d_ws;
  unsigned short* ws_img  = (unsigned short*)base;                 // I*R_PAD*D bf16
  base += (size_t)I_N * R_PAD * D_N * sizeof(unsigned short);
  unsigned short* ws_imgT = (unsigned short*)base;                 // I*D*K_R bf16
  base += (size_t)I_N * D_N * K_R * sizeof(unsigned short);
  unsigned short* ws_cap  = (unsigned short*)base;                 // C*W_PAD*D bf16
  base += (size_t)I_N * W_PAD * D_N * sizeof(unsigned short);
  float* ws_capnorm = (float*)base;                                // C*W f32
  base += (size_t)I_N * W_N * sizeof(float);
  float* ws_scores  = (float*)base;                                // I*C f32

  const int TB = 256;
  const long long n_img  = (long long)I_N * R_PAD * D_N;
  const long long n_imgT = (long long)I_N * D_N * K_R;
  const long long n_cap  = (long long)I_N * W_PAD * D_N;

  prep_img_bf16 <<<(unsigned)((n_img  + TB - 1) / TB), TB, 0, stream>>>(images, ws_img);
  prep_imgT_bf16<<<(unsigned)((n_imgT + TB - 1) / TB), TB, 0, stream>>>(images, ws_imgT);
  prep_cap_bf16 <<<(unsigned)((n_cap  + TB - 1) / TB), TB, 0, stream>>>(captions, ws_cap);
  prep_capnorm  <<<I_N, 32, 0, stream>>>(captions, ws_capnorm);

  scan_t2i_pair_kernel<<<dim3(I_N, I_N), 128, 0, stream>>>(
      ws_img, ws_imgT, ws_cap, captions, ws_capnorm, cap_lens, ws_scores);

  hinge_loss_kernel<<<1, 128, 0, stream>>>(ws_scores, (float*)d_out);
}
